// Decoder_68289980006849
// MI455X (gfx1250) — compile-verified, matrix operand order checked by reading, not run on hardware
//
#include <hip/hip_runtime.h>

// ---------------------------------------------------------------------------
// CDNA5 (gfx1250) transformer decoder:
//   bf16 WMMA GEMMs with async global->LDS double-buffered staging,
//   fused flash attention, bf16 activation pipeline (fp32 residual stream).
// ---------------------------------------------------------------------------

typedef __attribute__((ext_vector_type(8)))  float   v8f;
typedef __attribute__((ext_vector_type(16))) __bf16  bf16x16;
typedef __attribute__((ext_vector_type(8)))  __bf16  bf16x8;

#define D_MODEL 1024
#define N_HEAD  16
#define SEQ     512
#define BATCH   8
#define NROWS   (BATCH * SEQ)   // 4096
#define FFN     4096
#define NLAYER  3

__device__ __forceinline__ v8f vzero() {
  v8f z;
  #pragma unroll
  for (int i = 0; i < 8; ++i) z[i] = 0.0f;
  return z;
}

// A/B fragment gather for V_WMMA_*_16X16X32_BF16 from a row of >=32 bf16 in LDS.
// Frag elements 0..7  -> K = 8*half .. 8*half+7
//      elements 8..15 -> K = 16+8*half .. 16+8*half+7   => two ds_load_b128.
__device__ __forceinline__ bf16x16 frag_ld(const __bf16* p, int half) {
  union { bf16x16 v; bf16x8 h[2]; } u;
  u.h[0] = *(const bf16x8*)(p + 8 * half);
  u.h[1] = *(const bf16x8*)(p + 16 + 8 * half);
  return u.v;
}

__device__ __forceinline__ v8f wmma_bf16(bf16x16 a, bf16x16 b, v8f c) {
  return __builtin_amdgcn_wmma_f32_16x16x32_bf16(false, a, false, b,
                                                 (short)0, c, false, false);
}

// ---------------------------------------------------------------------------
// Async stage of one 128x64 bf16 tile pair (A and W) into LDS.
// 1024 16-byte chunks per tile; 256 threads -> 4 chunks each per matrix.
// Each wave issues exactly 8 async-copy instructions per call.
// ---------------------------------------------------------------------------
__device__ __forceinline__ void stage_tile_async(
    const __bf16* A, const __bf16* W, int bm, int bn, int K, int k0,
    __bf16* sAb, __bf16* sBb, int tid)
{
  const unsigned long long abase = (unsigned long long)A;
  const unsigned long long wbase = (unsigned long long)W;
  #pragma unroll
  for (int c = 0; c < 4; ++c) {
    const int idx = tid + c * 256;
    const int row = idx >> 3;
    const int col = (idx & 7) * 8;                      // bf16 elements
    const unsigned ga = (unsigned)(((bm + row) * K + k0 + col) * 2);
    const unsigned gw = (unsigned)(((bn + row) * K + k0 + col) * 2);
    const unsigned la = (unsigned)(uintptr_t)(sAb + row * 64 + col);
    const unsigned lw = (unsigned)(uintptr_t)(sBb + row * 64 + col);
    asm volatile("global_load_async_to_lds_b128 %0, %1, %2"
                 :: "v"(la), "v"(ga), "s"(abase) : "memory");
    asm volatile("global_load_async_to_lds_b128 %0, %1, %2"
                 :: "v"(lw), "v"(gw), "s"(wbase) : "memory");
  }
}

// ---------------------------------------------------------------------------
// GEMM: C[M,N] = A[M,K] @ W[N,K]^T + bias[N] (+ residual) (+ relu)
// A, W are bf16; accumulate fp32; output fp32 or bf16 (template).
// Block: 256 threads (8 waves), tile 128x128, BK=64, double-buffered LDS
// filled by GLOBAL_LOAD_ASYNC_TO_LDS_B128 (ASYNCcnt).
// ---------------------------------------------------------------------------
template<bool BF16OUT, bool RESID, bool RELU>
__global__ __launch_bounds__(256) void gemm_bf16_kernel(
    const __bf16* __restrict__ A, const __bf16* __restrict__ W,
    const float* __restrict__ bias, const float* __restrict__ resid,
    void* __restrict__ Cout, int M, int N, int K)
{
  __shared__ alignas(16) __bf16 sA[2][128 * 64];   // 2 x 16 KB
  __shared__ alignas(16) __bf16 sB[2][128 * 64];   // 2 x 16 KB

  const int tid  = threadIdx.x;
  const int wave = tid >> 5, lane = tid & 31;
  const int half = lane >> 4, r = lane & 15;
  const int bm = blockIdx.x * 128, bn = blockIdx.y * 128;
  const int wm = (wave & 1) * 64;        // 4 M-tiles of 16
  const int wn = (wave >> 1) * 32;       // 2 N-tiles of 16

  v8f acc[4][2];
  #pragma unroll
  for (int mt = 0; mt < 4; ++mt)
    #pragma unroll
    for (int nt = 0; nt < 2; ++nt) acc[mt][nt] = vzero();

  const int NT = K >> 6;                 // K / 64
  stage_tile_async(A, W, bm, bn, K, 0, sA[0], sB[0], tid);

  for (int kt = 0; kt < NT; ++kt) {
    const int cur = kt & 1;
    __syncthreads();                     // everyone done reading buf[cur^1]
    if (kt + 1 < NT) {
      stage_tile_async(A, W, bm, bn, K, (kt + 1) * 64, sA[cur ^ 1], sB[cur ^ 1], tid);
      asm volatile("s_wait_asynccnt 0x8" ::: "memory");  // tile kt landed
    } else {
      asm volatile("s_wait_asynccnt 0x0" ::: "memory");
    }
    __syncthreads();                     // tile kt visible block-wide

    #pragma unroll
    for (int kk = 0; kk < 2; ++kk) {     // two 32-wide WMMA K-steps
      bf16x16 af[4], bfr[2];
      #pragma unroll
      for (int mt = 0; mt < 4; ++mt)
        af[mt] = frag_ld(sA[cur] + (wm + mt * 16 + r) * 64 + kk * 32, half);
      #pragma unroll
      for (int nt = 0; nt < 2; ++nt)
        bfr[nt] = frag_ld(sB[cur] + (wn + nt * 16 + r) * 64 + kk * 32, half);
      #pragma unroll
      for (int mt = 0; mt < 4; ++mt)
        #pragma unroll
        for (int nt = 0; nt < 2; ++nt)
          acc[mt][nt] = wmma_bf16(af[mt], bfr[nt], acc[mt][nt]);
    }
  }

  // Epilogue. C/D layout: row = e + 8*half, col = lane&15.
  #pragma unroll
  for (int nt = 0; nt < 2; ++nt) {
    const int gn = bn + wn + nt * 16 + r;
    const float bb = bias[gn];
    #pragma unroll
    for (int mt = 0; mt < 4; ++mt) {
      size_t ro = (size_t)(bm + wm + mt * 16 + 8 * half) * N + gn;
      #pragma unroll
      for (int e = 0; e < 8; ++e, ro += N) {
        float v = acc[mt][nt][e] + bb;
        if (RESID) v += resid[ro];
        if (RELU)  v = fmaxf(v, 0.0f);
        if (BF16OUT) ((__bf16*)Cout)[ro] = (__bf16)v;
        else         ((float*)Cout)[ro]  = v;
      }
    }
  }
}

// ---------------------------------------------------------------------------
// Flash attention, per (batch, head, 64-query slab). 128 threads = 4 waves,
// each wave owns a 16-row Q tile. 32-key steps, online softmax, bf16 WMMA.
// Q/K/V are bf16 [NROWS, D_MODEL]; ctx written bf16.
// ---------------------------------------------------------------------------
__global__ __launch_bounds__(128) void flash_attn_kernel(
    const __bf16* __restrict__ Qb, const __bf16* __restrict__ Kb,
    const __bf16* __restrict__ Vb, __bf16* __restrict__ Ctx, int causal)
{
  __shared__ alignas(16) __bf16 sK[32 * 64];       // [key][d]
  __shared__ alignas(16) __bf16 sV[64 * 32];       // [d][key] (transposed)
  __shared__ alignas(16) __bf16 sP[4][16 * 32];    // per-wave P tile

  const int tid  = threadIdx.x;
  const int wave = tid >> 5, lane = tid & 31;
  const int half = lane >> 4, r = lane & 15;
  const int qblk = blockIdx.x, head = blockIdx.y, batch = blockIdx.z;
  const int hc   = head * 64;
  const int qrow = batch * SEQ + qblk * 64 + wave * 16;

  // Q tile (16 x 64) as two A-fragments, straight from global (bf16, aligned).
  bf16x16 qf[2];
  {
    const __bf16* qp = Qb + (size_t)(qrow + r) * D_MODEL + hc;
    #pragma unroll
    for (int f = 0; f < 2; ++f) {
      union { bf16x16 v; bf16x8 h[2]; } u;
      u.h[0] = *(const bf16x8*)(qp + f * 32 + 8 * half);
      u.h[1] = *(const bf16x8*)(qp + f * 32 + 16 + 8 * half);
      qf[f] = u.v;
    }
  }

  float m_run[8], l_run[8];
  #pragma unroll
  for (int e = 0; e < 8; ++e) { m_run[e] = -1e30f; l_run[e] = 0.0f; }
  v8f ctx[4];
  #pragma unroll
  for (int nt = 0; nt < 4; ++nt) ctx[nt] = vzero();

  const int q_hi = qblk * 64 + 63;                 // block-uniform

  for (int kt = 0; kt < SEQ / 32; ++kt) {
    const int kb = kt * 32;
    if (causal && kb > q_hi) break;                // uniform across block
    __syncthreads();
    // Cooperative staging: K row-major, V transposed. 2 chunks per thread.
    #pragma unroll
    for (int c = 0; c < 2; ++c) {
      const int idx = tid + c * 128;               // 256 8-elem chunks
      const int row = idx >> 3, col = (idx & 7) * 8;
      const __bf16* gp = Kb + (size_t)(batch * SEQ + kb + row) * D_MODEL + hc + col;
      *(bf16x8*)(sK + row * 64 + col) = *(const bf16x8*)gp;
      const bf16x8 vv = *(const bf16x8*)(Vb + (size_t)(batch * SEQ + kb + row) * D_MODEL + hc + col);
      #pragma unroll
      for (int j = 0; j < 8; ++j) sV[(col + j) * 32 + row] = vv[j];
    }
    __syncthreads();

    // S = Q @ K^T for 32 keys (two 16x16 tiles), DK=64 -> 2 K-steps each.
    v8f s0 = vzero(), s1 = vzero();
    #pragma unroll
    for (int f = 0; f < 2; ++f) {
      bf16x16 k0f = frag_ld(sK + r * 64        + f * 32, half);
      bf16x16 k1f = frag_ld(sK + (16 + r) * 64 + f * 32, half);
      s0 = wmma_bf16(qf[f], k0f, s0);
      s1 = wmma_bf16(qf[f], k1f, s1);
    }

    const int qbase = qblk * 64 + wave * 16;
    #pragma unroll
    for (int e = 0; e < 8; ++e) {
      const int qi = qbase + e + 8 * half;
      float v0 = s0[e] * 0.125f, v1 = s1[e] * 0.125f;
      if (causal) {
        if (kb + r      > qi) v0 = -1e9f;
        if (kb + 16 + r > qi) v1 = -1e9f;
      }
      float t = fmaxf(v0, v1);
      #pragma unroll
      for (int m = 1; m < 16; m <<= 1) t = fmaxf(t, __shfl_xor(t, m, 32));
      const float mnew = fmaxf(m_run[e], t);
      const float corr = __expf(m_run[e] - mnew);
      const float p0 = __expf(v0 - mnew);
      const float p1 = __expf(v1 - mnew);
      float rs = p0 + p1;
      #pragma unroll
      for (int m = 1; m < 16; m <<= 1) rs += __shfl_xor(rs, m, 32);
      l_run[e] = l_run[e] * corr + rs;
      m_run[e] = mnew;
      sP[wave][(e + 8 * half) * 32 + r]      = (__bf16)p0;
      sP[wave][(e + 8 * half) * 32 + 16 + r] = (__bf16)p1;
      #pragma unroll
      for (int nt = 0; nt < 4; ++nt) ctx[nt][e] *= corr;
    }
    __syncthreads();

    // ctx += P @ V
    bf16x16 pf = frag_ld(&sP[wave][r * 32], half);
    #pragma unroll
    for (int nt = 0; nt < 4; ++nt) {
      bf16x16 vf = frag_ld(sV + (nt * 16 + r) * 32, half);
      ctx[nt] = wmma_bf16(pf, vf, ctx[nt]);
    }
  }

  #pragma unroll
  for (int e = 0; e < 8; ++e) {
    const float inv = 1.0f / l_run[e];
    const int gm = qrow + e + 8 * half;
    #pragma unroll
    for (int nt = 0; nt < 4; ++nt)
      Ctx[(size_t)gm * D_MODEL + hc + nt * 16 + r] = (__bf16)(ctx[nt][e] * inv);
  }
}

// ---------------------------------------------------------------------------
// Elementwise kernels
// ---------------------------------------------------------------------------
__global__ __launch_bounds__(256) void embed_kernel(
    const int* __restrict__ tok, const float* __restrict__ emb,
    const float* __restrict__ pe, float* __restrict__ X)
{
  const int i = blockIdx.x * 256 + threadIdx.x;
  const int n = i >> 10, d = i & 1023;
  const int t = tok[n];
  X[i] = emb[(size_t)t * D_MODEL + d] * 32.0f
       + pe[(size_t)(n & (SEQ - 1)) * D_MODEL + d];
}

__global__ __launch_bounds__(256) void add_kernel(
    float* __restrict__ X, const float* __restrict__ Cb)
{
  const int i = blockIdx.x * 256 + threadIdx.x;
  X[i] += Cb[i];
}

__global__ __launch_bounds__(256) void cvt_kernel(
    const float* __restrict__ in, __bf16* __restrict__ out)
{
  const int i = (blockIdx.x * 256 + threadIdx.x) * 4;
  const float4 f = *(const float4*)(in + i);
  __bf16* o = out + i;
  o[0] = (__bf16)f.x; o[1] = (__bf16)f.y;
  o[2] = (__bf16)f.z; o[3] = (__bf16)f.w;
}

__global__ __launch_bounds__(256) void layernorm_kernel(
    const float* __restrict__ X, const float* __restrict__ gam,
    const float* __restrict__ bet, __bf16* __restrict__ Out)
{
  const int row = blockIdx.x;
  const float* x = X + (size_t)row * D_MODEL;
  float v[4], s = 0.0f, sq = 0.0f;
  #pragma unroll
  for (int i = 0; i < 4; ++i) {
    v[i] = x[threadIdx.x + i * 256];
    s += v[i]; sq += v[i] * v[i];
  }
  #pragma unroll
  for (int m = 1; m < 32; m <<= 1) {
    s  += __shfl_xor(s,  m, 32);
    sq += __shfl_xor(sq, m, 32);
  }
  __shared__ float red0[8], red1[8];
  __shared__ float smean, srstd;
  const int wv = threadIdx.x >> 5, lane = threadIdx.x & 31;
  if (lane == 0) { red0[wv] = s; red1[wv] = sq; }
  __syncthreads();
  if (threadIdx.x == 0) {
    float a = 0.0f, c = 0.0f;
    for (int w = 0; w < 8; ++w) { a += red0[w]; c += red1[w]; }
    const float mean = a * (1.0f / D_MODEL);
    const float var  = c * (1.0f / D_MODEL) - mean * mean;
    smean = mean; srstd = rsqrtf(var + 1e-5f);
  }
  __syncthreads();
  #pragma unroll
  for (int i = 0; i < 4; ++i) {
    const int d = threadIdx.x + i * 256;
    Out[(size_t)row * D_MODEL + d] =
        (__bf16)((v[i] - smean) * srstd * gam[d] + bet[d]);
  }
}

// ---------------------------------------------------------------------------
// Orchestration. Workspace (bytes), total ~92 MB:
//   X    fp32 [NROWS*D]      residual stream
//   Cb   fp32 [NROWS*D]      cross-attn output (saved for residual quirk)
//   Hb   bf16 [NROWS*D]      layernorm output
//   AB   bf16 [NROWS*FFN]    {Q,K,V,ctx} during attention / FFN hidden
//   encB bf16 [NROWS*D]      enc_output converted once
//   Wst  bf16 [FFN*D]        per-GEMM weight staging (converted fp32->bf16)
// ---------------------------------------------------------------------------
extern "C" void kernel_launch(void* const* d_in, const int* in_sizes, int n_in,
                              void* d_out, int out_size, void* d_ws, size_t ws_size,
                              hipStream_t stream) {
  (void)in_sizes; (void)n_in; (void)out_size; (void)ws_size;
  const int*   dec = (const int*)d_in[0];
  const float* enc = (const float*)d_in[2];
  const float* emb = (const float*)d_in[4];
  const float* pe  = (const float*)d_in[5];
  const float* wq  = (const float*)d_in[6];
  const float* bq  = (const float*)d_in[7];
  const float* wk  = (const float*)d_in[8];
  const float* bk  = (const float*)d_in[9];
  const float* wv  = (const float*)d_in[10];
  const float* bv  = (const float*)d_in[11];
  const float* wo  = (const float*)d_in[12];
  const float* bo  = (const float*)d_in[13];
  const float* fw1 = (const float*)d_in[14];
  const float* fb1 = (const float*)d_in[15];
  const float* fw2 = (const float*)d_in[16];
  const float* fb2 = (const float*)d_in[17];
  const float* lng = (const float*)d_in[18];
  const float* lnb = (const float*)d_in[19];
  float* out = (float*)d_out;

  const size_t ND = (size_t)NROWS * D_MODEL;
  char* w = (char*)d_ws;
  float*  X    = (float*)w;   w += ND * 4;
  float*  Cb   = (float*)w;   w += ND * 4;
  __bf16* Hb   = (__bf16*)w;  w += ND * 2;
  __bf16* AB   = (__bf16*)w;  w += (size_t)NROWS * FFN * 2;
  __bf16* encB = (__bf16*)w;  w += ND * 2;
  __bf16* Wst  = (__bf16*)w;  w += (size_t)FFN * D_MODEL * 2;
  __bf16* Qs = AB, *Ks = AB + ND, *Vs = AB + 2 * ND, *Cx = AB + 3 * ND;

  const int elemBlocks = (int)(ND / 256);
  const int DD_BLOCKS = D_MODEL * D_MODEL / 1024;      // cvt blocks for DxD
  const int FD_BLOCKS = FFN * D_MODEL / 1024;          // cvt blocks for FxD

  auto cvt = [&](const float* src, __bf16* dst, int blocks) {
    cvt_kernel<<<blocks, 256, 0, stream>>>(src, dst);
  };
  #define GEMM(BO, RS, RL, A_, W_, B_, R_, C_, M_, N_, K_)                      \
    gemm_bf16_kernel<BO, RS, RL><<<dim3((M_) / 128, (N_) / 128), 256, 0,        \
                                   stream>>>((A_), (W_), (B_), (R_), (C_),      \
                                             (M_), (N_), (K_))

  embed_kernel<<<elemBlocks, 256, 0, stream>>>(dec, emb, pe, X);
  cvt(enc, encB, (int)(ND / 1024));

  for (int l = 0; l < NLAYER; ++l) {
    const size_t a0  = ((size_t)l * 2 + 0) * D_MODEL * D_MODEL;
    const size_t a1  = ((size_t)l * 2 + 1) * D_MODEL * D_MODEL;
    const size_t ab0 = ((size_t)l * 2 + 0) * D_MODEL;
    const size_t ab1 = ((size_t)l * 2 + 1) * D_MODEL;
    const float* g0 = lng + ((size_t)l * 3 + 0) * D_MODEL;
    const float* b0 = lnb + ((size_t)l * 3 + 0) * D_MODEL;
    const float* g1 = lng + ((size_t)l * 3 + 1) * D_MODEL;
    const float* b1 = lnb + ((size_t)l * 3 + 1) * D_MODEL;
    const float* g2 = lng + ((size_t)l * 3 + 2) * D_MODEL;
    const float* b2 = lnb + ((size_t)l * 3 + 2) * D_MODEL;

    // ---- masked self-attention ----
    layernorm_kernel<<<NROWS, 256, 0, stream>>>(X, g0, b0, Hb);
    cvt(wq + a0, Wst, DD_BLOCKS);
    GEMM(true, false, false, Hb, Wst, bq + ab0, nullptr, Qs, NROWS, D_MODEL, D_MODEL);
    cvt(wk + a0, Wst, DD_BLOCKS);
    GEMM(true, false, false, Hb, Wst, bk + ab0, nullptr, Ks, NROWS, D_MODEL, D_MODEL);
    cvt(wv + a0, Wst, DD_BLOCKS);
    GEMM(true, false, false, Hb, Wst, bv + ab0, nullptr, Vs, NROWS, D_MODEL, D_MODEL);
    flash_attn_kernel<<<dim3(SEQ / 64, N_HEAD, BATCH), 128, 0, stream>>>(
        Qs, Ks, Vs, Cx, 1);
    cvt(wo + a0, Wst, DD_BLOCKS);
    GEMM(false, true, false, Cx, Wst, bo + ab0, X, X, NROWS, D_MODEL, D_MODEL);

    // ---- encoder-decoder attention ----
    layernorm_kernel<<<NROWS, 256, 0, stream>>>(X, g1, b1, Hb);
    cvt(wq + a1, Wst, DD_BLOCKS);
    GEMM(true, false, false, Hb, Wst, bq + ab1, nullptr, Qs, NROWS, D_MODEL, D_MODEL);
    cvt(wk + a1, Wst, DD_BLOCKS);
    GEMM(true, false, false, encB, Wst, bk + ab1, nullptr, Ks, NROWS, D_MODEL, D_MODEL);
    cvt(wv + a1, Wst, DD_BLOCKS);
    GEMM(true, false, false, encB, Wst, bv + ab1, nullptr, Vs, NROWS, D_MODEL, D_MODEL);
    flash_attn_kernel<<<dim3(SEQ / 64, N_HEAD, BATCH), 128, 0, stream>>>(
        Qs, Ks, Vs, Cx, 0);
    cvt(wo + a1, Wst, DD_BLOCKS);
    GEMM(false, false, false, Cx, Wst, bo + ab1, nullptr, Cb, NROWS, D_MODEL, D_MODEL);
    add_kernel<<<elemBlocks, 256, 0, stream>>>(X, Cb);

    // ---- FFN (reference quirk: residual is cross-attn output c, not x) ----
    layernorm_kernel<<<NROWS, 256, 0, stream>>>(X, g2, b2, Hb);
    cvt(fw1 + (size_t)l * FFN * D_MODEL, Wst, FD_BLOCKS);
    GEMM(true, false, true, Hb, Wst, fb1 + (size_t)l * FFN, nullptr, AB,
         NROWS, FFN, D_MODEL);
    cvt(fw2 + (size_t)l * D_MODEL * FFN, Wst, FD_BLOCKS);
    void* dst = (l == NLAYER - 1) ? (void*)out : (void*)X;
    GEMM(false, true, false, AB, Wst, fb2 + (size_t)l * D_MODEL, Cb, dst,
         NROWS, D_MODEL, FFN);
  }
  #undef GEMM
}
